// MosaicMemory_49314814493257
// MI455X (gfx1250) — matrix operand order repaired and checked by reading, not compiled, not run on hardware
//
#include <hip/hip_runtime.h>
#include <stdint.h>

// ---------------- problem dims ----------------
#define B_  4
#define S_  4096
#define D_  1024
#define H_  4
#define NB_ 4096
#define A_  4
#define DK_ 64
#define DV_ 128
#define DT_ 32
#define BITS_ 12
#define N_  (B_ * S_)          // 16384 tokens
#define PCOLS 256              // [qkey 64 | wkey 64 | wval 128]
#define EPS_ 1e-6f
#define ETA_ 0.1f

// padded LDS strides produced by the TDM pad feature (f32 staging tiles)
#define USTRIDE 1028           // 1024 + 4 pads of 1 dword (interval 256)
#define UIDX(r, c) ((r) * USTRIDE + (c) + ((c) >> 8))
#define RSTRIDE 130            // 128 + 1 pad of 2 dwords (interval 128)
// packed-bf16 tile strides (dwords = bf16 pairs per row + pad)
#define UBSTR 516              // 512 pairs + 4  -> bank 4*lane, conflict-free
#define RBSTR 65               // 64 pairs + 1   -> bank 1*lane, conflict-free

typedef __attribute__((ext_vector_type(16))) __bf16 v16bf;
typedef __attribute__((ext_vector_type(8)))  float v8f;
typedef __attribute__((ext_vector_type(2)))  __bf16 bf16x2;
typedef __attribute__((ext_vector_type(4)))  unsigned tdmU4;
typedef __attribute__((ext_vector_type(8)))  int      tdmI8;
typedef __attribute__((ext_vector_type(4)))  int      tdmI4;

union FragBF { v16bf v; unsigned u[8]; };

#if __has_builtin(__builtin_amdgcn_tensor_load_to_lds)
#define HAVE_TDM 1
#else
#define HAVE_TDM 0
#endif

// pack two f32 -> packed bf16x2 dword; prefer hardware cvt, else let clang
// lower the fptrunc (RNE) -- both beat a hand-rolled integer sequence.
static __device__ __forceinline__ unsigned pk_bf16(float lo, float hi) {
#if __has_builtin(__builtin_amdgcn_cvt_pk_bf16_f32)
  bf16x2 p = __builtin_amdgcn_cvt_pk_bf16_f32(lo, hi);
#else
  bf16x2 p = { (__bf16)lo, (__bf16)hi };
#endif
  unsigned r; __builtin_memcpy(&r, &p, 4);
  return r;
}

static __device__ __forceinline__ unsigned lds_off(const void* p) {
  // LDS aperture: low 32 bits of the flat address are the LDS byte offset
  return (unsigned)(uintptr_t)p;
}

#if HAVE_TDM
// Issue one TDM 2D tile load (D# groups built per CDNA5 ISA 8.3/8.4).
// cfg = group1 dword0: data_size=4B | pad_enable | pad_interval | pad_amount.
static __device__ __forceinline__ void tdm_load_2d(
    unsigned lds_addr, const void* gptr,
    unsigned tensor_d0, unsigned tensor_d1,
    unsigned tile_d0, unsigned tile_d1,
    unsigned stride0, unsigned cfg)
{
  const unsigned long long ga = (unsigned long long)(uintptr_t)gptr;
  tdmU4 g0;
  g0[0] = 1u;                                        // count=1, user mode
  g0[1] = lds_addr;                                  // LDS dest (bytes)
  g0[2] = (unsigned)(ga & 0xFFFFFFFFu);              // global_addr lo
  g0[3] = (unsigned)((ga >> 32) & 0x1FFFFFFu)        // global_addr[56:32]
        | 0x80000000u;                               // type=2 ("image")
  tdmI8 g1;
  g1[0] = (int)cfg;                                  // wg_mask=0 | data_size | pad cfg
  g1[1] = (int)((tensor_d0 & 0xFFFFu) << 16);        // tensor_dim0[15:0]
  g1[2] = (int)(((tensor_d0 >> 16) & 0xFFFFu) |      // tensor_dim0[31:16]
                ((tensor_d1 & 0xFFFFu) << 16));      // tensor_dim1[15:0]
  g1[3] = (int)(((tensor_d1 >> 16) & 0xFFFFu) |      // tensor_dim1[31:16]
                (tile_d0 << 16));                    // tile_dim0
  g1[4] = (int)(tile_d1 & 0xFFFFu);                  // tile_dim1 (tile_dim2=0)
  g1[5] = (int)stride0;                              // tensor_dim0_stride lo32
  g1[6] = 0;                                         // stride0 hi16 | stride1 lo16
  g1[7] = 0;                                         // stride1 hi32
  const tdmI4 z4 = {0, 0, 0, 0};                     // dims 2/3 unused (2D tile)
#if defined(__clang_major__) && __clang_major__ >= 23
  const tdmI8 z8 = {0, 0, 0, 0, 0, 0, 0, 0};
  __builtin_amdgcn_tensor_load_to_lds(g0, g1, z4, z4, z8, 0);
#else
  __builtin_amdgcn_tensor_load_to_lds(g0, g1, z4, z4, 0);
#endif
}
#endif

// =====================================================================
// Kernel 1: fused projection GEMM  [N,1024] x [1024,256] -> proj
//           (qkey | wkey | wval) via v_wmma_f32_16x16x32_bf16.
// TDM stages the f32 u tile in LDS; one block-wide pass packs it into a
// bf16x2 tile so A-fragments are pure ds_load_b32 in the K loop. Gate
// GEMV runs in f32 from the staging tile.
// =====================================================================
__global__ __launch_bounds__(256)
void proj_kernel(const float* __restrict__ u,
                 const float* __restrict__ Wq,
                 const float* __restrict__ Wwk,
                 const float* __restrict__ Wv,
                 const float* __restrict__ Wg,
                 const float* __restrict__ bg,
                 float* __restrict__ proj,
                 float* __restrict__ gatebuf)
{
  __shared__ float    uF[16 * USTRIDE];   // f32 staging (TDM dest)
  __shared__ unsigned uB[16 * UBSTR];     // packed bf16 pairs

  const int tid = threadIdx.x;
  const int rowBase = blockIdx.x * 16;

#if HAVE_TDM
  if (tid < 32) {   // one wave programs the TDM, waits on TENSORcnt
    // cfg: data_size=4B(2<<16) | pad_en(1<<20) | interval=256dw(7<<22) | amount=1dw(0<<25)
    tdm_load_2d(lds_off(uF), u + (size_t)rowBase * D_,
                D_, 16, D_, 16, D_, (2u << 16) | (1u << 20) | (7u << 22));
    __builtin_amdgcn_s_wait_tensorcnt(0);
  }
#else
  for (int i = tid; i < 16 * D_; i += 256) {
    int r = i >> 10, c = i & (D_ - 1);
    uF[UIDX(r, c)] = u[(size_t)(rowBase + r) * D_ + c];
  }
#endif
  __syncthreads();

  // one-shot pack: f32 tile -> bf16x2 tile (A-side cvt hoisted out of K loop)
  for (int i = tid; i < 16 * (D_ / 2); i += 256) {
    const int r = i >> 9, cp = i & 511, c = cp * 2;
    uB[r * UBSTR + cp] = pk_bf16(uF[UIDX(r, c)], uF[UIDX(r, c + 1)]);
  }
  __syncthreads();

  const int wave = tid >> 5, lane = tid & 31;
  const int mrow = lane & 15, half = lane >> 4;

  for (int t = 0; t < 2; ++t) {
    const int colBase = (wave * 2 + t) * 16;
    const int col = colBase + mrow;                 // B-matrix column this lane feeds
    const float* wrow;
    if (col < 64)        wrow = Wq  + (size_t)col * D_;
    else if (col < 128)  wrow = Wwk + (size_t)(col - 64) * D_;
    else                 wrow = Wv  + (size_t)(col - 128) * D_;

    v8f acc = {0.f, 0.f, 0.f, 0.f, 0.f, 0.f, 0.f, 0.f};
    for (int kk = 0; kk < D_; kk += 32) {
      __builtin_prefetch(wrow + kk + 64, 0, 1);     // global_prefetch_b8
      FragBF a, b;
      // A: 16x32 bf16 (lanes 0-15: M rows, K 0-7/16-23; lanes 16-31: K 8-15/24-31)
      #pragma unroll
      for (int i = 0; i < 8; ++i) {
        const int c = kk + ((i >> 2) << 4) + (half << 3) + ((i & 3) << 1);
        a.u[i] = uB[mrow * UBSTR + (c >> 1)];       // plain ds_load_b32
      }
      // B: 32x16 bf16 (lanes 0-15: K 0-7/16-23; lanes 16-31: K 8-15/24-31)
      #pragma unroll
      for (int v = 0; v < 8; ++v) {
        const float2 p = *(const float2*)(wrow + kk + ((v >> 2) << 4) + (half << 3) + ((v & 3) << 1));
        b.u[v] = pk_bf16(p.x, p.y);
      }
      acc = __builtin_amdgcn_wmma_f32_16x16x32_bf16(
          false, a.v, false, b.v, (short)0, acc, false, false);
    }
    // C/D layout: VGPR v -> M = v (lanes 0-15) or v+8 (lanes 16-31), N = lane&15
    #pragma unroll
    for (int v = 0; v < 8; ++v) {
      const int r = rowBase + v + half * 8;
      proj[(size_t)r * PCOLS + colBase + mrow] = acc[v];
    }
  }

  // gate logit (f32 from staging tile): each wave -> 2 rows
  for (int t = 0; t < 2; ++t) {
    const int r16 = wave * 2 + t;
    float s = 0.f;
    for (int k = lane; k < D_; k += 32) s += uF[UIDX(r16, k)] * Wg[k];
    s += __shfl_xor(s, 16, 32);
    s += __shfl_xor(s,  8, 32);
    s += __shfl_xor(s,  4, 32);
    s += __shfl_xor(s,  2, 32);
    s += __shfl_xor(s,  1, 32);
    if (lane == 0) gatebuf[rowBase + r16] = s + bg[0];
  }
}

// =====================================================================
// Kernel 2: routing + gather + softmax read + scatter update.
// One wave per token, 8 tokens per block. R/Wvsa tables pulled into LDS
// by the TDM. Gathers hit L2 (memK+memV+memT = 56 MB < 192 MB).
// =====================================================================
__global__ __launch_bounds__(256)
void token_kernel(const float* __restrict__ proj,
                  const float* __restrict__ gatebuf,
                  const float* __restrict__ memK,
                  const float* __restrict__ memV,
                  const float* __restrict__ memT,
                  const float* __restrict__ R,     // [H,DK,BITS]
                  const float* __restrict__ Wvsa,  // [DK,DT]
                  float* __restrict__ readb,       // [N,DV]
                  float* __restrict__ Knew,
                  float* __restrict__ Vnew,
                  float* __restrict__ Tnew)
{
  __shared__ float Rs[H_ * DK_ * BITS_];   // h*768 + k*12 + b
  __shared__ float Ws[DK_ * DT_];          // k*32 + t
  __shared__ float qS[8][DK_], wkS[8][DK_];
  __shared__ float qtS[8][DT_], wtS[8][DT_];
  __shared__ float simS[8][16], tsS[8][16], wS[8][16];
  __shared__ float effS[8][H_];
  __shared__ int   asS[8][H_];

  const int tid = threadIdx.x, wave = tid >> 5, lane = tid & 31;
  const int n = blockIdx.x * 8 + wave;

#if HAVE_TDM
  if (tid < 32) {   // 1D table copies via TDM (no padding)
    tdm_load_2d(lds_off(Rs), R,    H_ * DK_ * BITS_, 1, H_ * DK_ * BITS_, 1,
                H_ * DK_ * BITS_, (2u << 16));
    tdm_load_2d(lds_off(Ws), Wvsa, DK_ * DT_, 1, DK_ * DT_, 1,
                DK_ * DT_, (2u << 16));
    __builtin_amdgcn_s_wait_tensorcnt(0);
  }
#else
  for (int i = tid; i < H_ * DK_ * BITS_; i += 256) Rs[i] = R[i];
  for (int i = tid; i < DK_ * DT_;        i += 256) Ws[i] = Wvsa[i];
#endif

  const float* prow = proj + (size_t)n * PCOLS;
  qS[wave][lane]       = prow[lane];
  qS[wave][lane + 32]  = prow[lane + 32];
  wkS[wave][lane]      = prow[64 + lane];
  wkS[wave][lane + 32] = prow[96 + lane];
  __syncthreads();

  const float gate  = 1.f / (1.f + __expf(-gatebuf[n]));
  const float wmask = (gate > 0.5f) ? 1.f : 0.f;

  // ---- LSH routing: lane b (<12) computes bit b; wave32 ballot packs 2^b ----
  int bidx[H_];
  #pragma unroll
  for (int h = 0; h < H_; ++h) {
    float s = 0.f;
    if (lane < BITS_) {
      const float* rp = Rs + h * (DK_ * BITS_) + lane;
      #pragma unroll 8
      for (int k = 0; k < DK_; ++k) s += qS[wave][k] * rp[k * BITS_];
    }
    unsigned m = __builtin_amdgcn_ballot_w32(lane < BITS_ && s > 0.f);
    bidx[h] = (int)(m & 0xFFFu);
  }

  // ---- tags: lane t computes qtag[t], wtag[t] ----
  float wt;
  {
    float aq = 0.f, aw = 0.f;
    #pragma unroll 8
    for (int k = 0; k < DK_; ++k) {
      const float wv = Ws[k * DT_ + lane];
      aq += qS[wave][k] * wv;
      aw += wkS[wave][k] * wv;
    }
    qtS[wave][lane] = tanhf(aq);
    wt = tanhf(aw);
    wtS[wave][lane] = wt;
  }

  // ---- l2 norms of wkey / wtag (wave shuffle reductions) ----
  float s2 = wkS[wave][lane] * wkS[wave][lane]
           + wkS[wave][lane + 32] * wkS[wave][lane + 32];
  s2 += __shfl_xor(s2, 16, 32); s2 += __shfl_xor(s2, 8, 32);
  s2 += __shfl_xor(s2, 4, 32);  s2 += __shfl_xor(s2, 2, 32);
  s2 += __shfl_xor(s2, 1, 32);
  const float inv_wk = 1.f / (sqrtf(s2) + EPS_);

  float t2 = wt * wt;
  t2 += __shfl_xor(t2, 16, 32); t2 += __shfl_xor(t2, 8, 32);
  t2 += __shfl_xor(t2, 4, 32);  t2 += __shfl_xor(t2, 2, 32);
  t2 += __shfl_xor(t2, 1, 32);
  const float inv_wt = 1.f / (sqrtf(t2) + EPS_);
  __syncthreads();   // qtS / wtS visible cross-lane

  // ---- per-slot scores, cosine sims (lane s < 16 handles slot s) ----
  float myscore = 0.f;
  if (lane < 16) {
    const int h = lane >> 2, a = lane & 3;
    const int base = ((h * NB_ + bidx[h]) * A_ + a);
    const float* kp = memK + (size_t)base * DK_;
    float kq = 0.f, kw = 0.f, kn = 0.f;
    #pragma unroll 8
    for (int k = 0; k < DK_; ++k) {
      const float kv = kp[k];
      kq += qS[wave][k] * kv; kw += wkS[wave][k] * kv; kn += kv * kv;
    }
    const float* tp = memT + (size_t)base * DT_;
    float tq = 0.f, tw = 0.f, tn = 0.f;
    #pragma unroll 8
    for (int t = 0; t < DT_; ++t) {
      const float tv = tp[t];
      tq += qtS[wave][t] * tv; tw += wtS[wave][t] * tv; tn += tv * tv;
    }
    myscore = kq * 0.125f + tq * 0.17677669529663688f;  // /sqrt(64), /sqrt(32)
    simS[wave][lane] = kw * inv_wk / (sqrtf(kn) + EPS_);
    tsS[wave][lane]  = tw * inv_wt / (sqrtf(tn) + EPS_);
  }
  // softmax over the 16 slots (low half-wave, width-16 shuffles)
  float mx = myscore;
  mx = fmaxf(mx, __shfl_xor(mx, 8, 16));
  mx = fmaxf(mx, __shfl_xor(mx, 4, 16));
  mx = fmaxf(mx, __shfl_xor(mx, 2, 16));
  mx = fmaxf(mx, __shfl_xor(mx, 1, 16));
  float e = (lane < 16) ? __expf(myscore - mx) : 0.f;
  float se = e;
  se += __shfl_xor(se, 8, 16); se += __shfl_xor(se, 4, 16);
  se += __shfl_xor(se, 2, 16); se += __shfl_xor(se, 1, 16);
  if (lane < 16) wS[wave][lane] = e / se;

  // eff / argmax per hash (lanes < 4)
  if (lane < H_) {
    const int h = lane;
    float best = -1e30f, bt = -1e30f; int ba = 0;
    #pragma unroll
    for (int a = 0; a < A_; ++a) {
      const float sv = simS[wave][h * A_ + a];
      if (sv > best) { best = sv; ba = a; }
      bt = fmaxf(bt, tsS[wave][h * A_ + a]);
    }
    const float nov = 1.f - bt;
    effS[wave][h] = ETA_ * gate * wmask
                  * (best >= 0.f ? 1.f : 0.f)
                  * (nov  >= 0.f ? 1.f : 0.f);
    asS[wave][h] = ba;
  }
  __syncthreads();   // wS / effS / asS visible

  // ---- weighted read: lane covers dims lane+32j ----
  float racc[4] = {0.f, 0.f, 0.f, 0.f};
  for (int s = 0; s < 16; ++s) {
    const int h = s >> 2, a = s & 3;
    const int base = ((h * NB_ + bidx[h]) * A_ + a);
    const float* vp = memV + (size_t)base * DV_;
    const float ws = wS[wave][s];
    #pragma unroll
    for (int j = 0; j < 4; ++j) racc[j] += ws * vp[lane + 32 * j];
  }
  #pragma unroll
  for (int j = 0; j < 4; ++j) readb[(size_t)n * DV_ + lane + 32 * j] = racc[j];

  // ---- gated scatter write (f32 atomics into pre-copied outputs) ----
  #pragma unroll
  for (int h = 0; h < H_; ++h) {
    const float ef = effS[wave][h];
    if (ef != 0.f) {
      const int base = ((h * NB_ + bidx[h]) * A_ + asS[wave][h]);
      float* kd = Knew + (size_t)base * DK_;
      const float* ko = memK + (size_t)base * DK_;
      #pragma unroll
      for (int j = 0; j < 2; ++j) {
        const int d = lane + 32 * j;
        atomicAdd(kd + d, ef * (wkS[wave][d] - ko[d]));
      }
      float* vd = Vnew + (size_t)base * DV_;
      const float* vo = memV + (size_t)base * DV_;
      #pragma unroll
      for (int j = 0; j < 4; ++j) {
        const int d = lane + 32 * j;
        atomicAdd(vd + d, ef * (prow[128 + d] - vo[d]));
      }
      float* td = Tnew + (size_t)base * DT_;
      const float* to = memT + (size_t)base * DT_;
      atomicAdd(td + lane, ef * (wtS[wave][lane] - to[lane]));
    }
  }
}

// =====================================================================
// Kernel 3: y = read @ Wo^T   [N,128] x [128,1024] via WMMA bf16.
// TDM stages the f32 read tile; one pass packs it to bf16x2; fragments
// are then pure ds_load_b32 across all 8 column tiles per wave.
// =====================================================================
__global__ __launch_bounds__(256)
void outproj_kernel(const float* __restrict__ readb,
                    const float* __restrict__ Wo,   // [D, DV]
                    float* __restrict__ y)
{
  __shared__ float    rT[16 * RSTRIDE];   // f32 staging (TDM dest)
  __shared__ unsigned rB[16 * RBSTR];     // packed bf16 pairs

  const int tid = threadIdx.x;
  const int rowBase = blockIdx.x * 16;

#if HAVE_TDM
  if (tid < 32) {
    // cfg: 4B | pad_en | interval=128dw(6<<22) | amount=2dw(1<<25)
    tdm_load_2d(lds_off(rT), readb + (size_t)rowBase * DV_,
                DV_, 16, DV_, 16, DV_,
                (2u << 16) | (1u << 20) | (6u << 22) | (1u << 25));
    __builtin_amdgcn_s_wait_tensorcnt(0);
  }
#else
  for (int i = tid; i < 16 * DV_; i += 256) {
    int r = i >> 7, c = i & (DV_ - 1);
    rT[r * RSTRIDE + c] = readb[(size_t)(rowBase + r) * DV_ + c];
  }
#endif
  __syncthreads();

  for (int i = tid; i < 16 * (DV_ / 2); i += 256) {
    const int r = i >> 6, cp = i & 63, c = cp * 2;
    rB[r * RBSTR + cp] = pk_bf16(rT[r * RSTRIDE + c], rT[r * RSTRIDE + c + 1]);
  }
  __syncthreads();

  const int wave = tid >> 5, lane = tid & 31;
  const int mrow = lane & 15, half = lane >> 4;

  for (int t = 0; t < 8; ++t) {
    const int colBase = (wave * 8 + t) * 16;
    const float* wrow = Wo + (size_t)(colBase + mrow) * DV_;
    __builtin_prefetch(wrow, 0, 1);

    v8f acc = {0.f, 0.f, 0.f, 0.f, 0.f, 0.f, 0.f, 0.f};
    for (int kk = 0; kk < DV_; kk += 32) {
      FragBF a, b;
      #pragma unroll
      for (int i = 0; i < 8; ++i) {
        const int c = kk + ((i >> 2) << 4) + (half << 3) + ((i & 3) << 1);
        a.u[i] = rB[mrow * RBSTR + (c >> 1)];       // plain ds_load_b32
      }
      #pragma unroll
      for (int v = 0; v < 8; ++v) {
        const float2 p = *(const float2*)(wrow + kk + ((v >> 2) << 4) + (half << 3) + ((v & 3) << 1));
        b.u[v] = pk_bf16(p.x, p.y);
      }
      acc = __builtin_amdgcn_wmma_f32_16x16x32_bf16(
          false, a.v, false, b.v, (short)0, acc, false, false);
    }
    #pragma unroll
    for (int v = 0; v < 8; ++v) {
      const int r = rowBase + v + half * 8;
      y[(size_t)r * D_ + colBase + mrow] = acc[v];
    }
  }
}

// =====================================================================
extern "C" void kernel_launch(void* const* d_in, const int* in_sizes, int n_in,
                              void* d_out, int out_size, void* d_ws, size_t ws_size,
                              hipStream_t stream) {
  const float* u    = (const float*)d_in[0];
  const float* memK = (const float*)d_in[1];
  const float* memV = (const float*)d_in[2];
  const float* memT = (const float*)d_in[3];
  const float* Wq   = (const float*)d_in[4];
  const float* Wwk  = (const float*)d_in[5];
  const float* Wv   = (const float*)d_in[6];
  const float* Wo   = (const float*)d_in[7];
  const float* Wg   = (const float*)d_in[8];
  const float* bg   = (const float*)d_in[9];
  const float* Wvsa = (const float*)d_in[10];
  const float* R    = (const float*)d_in[11];

  // output layout: y | Knew | Vnew | Tnew (flat, in return order)
  float* y    = (float*)d_out;
  float* Knew = y + (size_t)N_ * D_;
  float* Vnew = Knew + (size_t)H_ * NB_ * A_ * DK_;
  float* Tnew = Vnew + (size_t)H_ * NB_ * A_ * DV_;

  // workspace: proj [N,256] | gate [N] | read [N,128]
  float* proj    = (float*)d_ws;
  float* gatebuf = proj + (size_t)N_ * PCOLS;
  float* readb   = gatebuf + N_;

  // seed output tables with originals; token_kernel adds deltas atomically
  hipMemcpyAsync(Knew, memK, (size_t)H_ * NB_ * A_ * DK_ * sizeof(float),
                 hipMemcpyDeviceToDevice, stream);
  hipMemcpyAsync(Vnew, memV, (size_t)H_ * NB_ * A_ * DV_ * sizeof(float),
                 hipMemcpyDeviceToDevice, stream);
  hipMemcpyAsync(Tnew, memT, (size_t)H_ * NB_ * A_ * DT_ * sizeof(float),
                 hipMemcpyDeviceToDevice, stream);

  proj_kernel<<<N_ / 16, 256, 0, stream>>>(u, Wq, Wwk, Wv, Wg, bg, proj, gatebuf);
  token_kernel<<<N_ / 8, 256, 0, stream>>>(proj, gatebuf, memK, memV, memT,
                                           R, Wvsa, readb, Knew, Vnew, Tnew);
  outproj_kernel<<<N_ / 16, 256, 0, stream>>>(readb, Wo, y);
}
